// EncoderBlock_46420006535529
// MI455X (gfx1250) — compile-verified, matrix operand order checked by reading, not run on hardware
//
#include <hip/hip_runtime.h>
#include <hip/hip_bf16.h>

typedef __bf16 bf16;
typedef __attribute__((ext_vector_type(16))) __bf16 v16bf;
typedef __attribute__((ext_vector_type(8)))  __bf16 v8bf;
typedef __attribute__((ext_vector_type(8)))  float  v8f;

#define NPOS 36864   // 192*192
#define NAX  192
#define DMODEL 128
#define NHEAD 4
#define DHEAD 32

// ---------------------------------------------------------------------------
// WMMA + async-copy helpers
// ---------------------------------------------------------------------------
__device__ __forceinline__ v8f wmma_bf16(v16bf a, v16bf b, v8f c) {
  return __builtin_amdgcn_wmma_f32_16x16x32_bf16(false, a, false, b, (short)0, c,
                                                 false, false);
}

// Load a 16x32 (MxK) bf16 fragment in the A layout from row-major storage.
// lanes 0-15: row=lane, K = {0..7, 16..23}; lanes 16-31: row=lane-16, K={8..15, 24..31}
__device__ __forceinline__ v16bf load_frag(const bf16* base, int ld) {
  int lane = threadIdx.x & 31;
  const bf16* p = base + (size_t)(lane & 15) * ld + ((lane & 16) >> 1);
  v8bf lo = *(const v8bf*)(p);
  v8bf hi = *(const v8bf*)(p + 16);
  return __builtin_shufflevector(lo, hi, 0, 1, 2, 3, 4, 5, 6, 7,
                                 8, 9, 10, 11, 12, 13, 14, 15);
}

// Wave-relative LDS byte offset of a shared-memory pointer (async-LDS VDST operand).
__device__ __forceinline__ unsigned lds_off(const void* p) {
  return (unsigned)(unsigned long long)
      (__attribute__((address_space(3))) const void*)p;
}

// Direct global -> LDS async copy of 16 bytes per lane (ASYNCcnt-tracked).
__device__ __forceinline__ void async_copy_b128(const void* gaddr, unsigned ldsaddr) {
  asm volatile("global_load_async_to_lds_b128 %0, %1, off"
               :: "v"(ldsaddr), "v"(gaddr)
               : "memory");
}

__device__ __forceinline__ void wait_async0() {
  asm volatile("s_wait_asynccnt 0x0" ::: "memory");
}

// ---------------------------------------------------------------------------
// Weight transpose + convert: in f32 (R x C) row-major -> out bf16 (C x R)
// ---------------------------------------------------------------------------
__global__ void wconv_kernel(const float* __restrict__ w, bf16* __restrict__ wt,
                             int R, int C) {
  int idx = blockIdx.x * 256 + threadIdx.x;
  if (idx < R * C) {
    int r = idx / C, c = idx % C;
    wt[(size_t)c * R + r] = (bf16)w[idx];
  }
}

// ---------------------------------------------------------------------------
// LayerNorm (+ optional attention pair-bias x.wb) kernel.
// One block (128 threads) per position p. Writes bf16 xn, optionally permuted
// (transpose=1 stores row p=(i*192+j) at j*192+i for the column-axial pass).
// ---------------------------------------------------------------------------
__global__ void __launch_bounds__(128)
ln_kernel(const float* __restrict__ x, const float* __restrict__ lng,
          const float* __restrict__ lnb, const float* __restrict__ wb,
          bf16* __restrict__ xn, float* __restrict__ biasb,
          int transpose, int doBias) {
  int p = blockIdx.x;
  int d = threadIdx.x;
  float v = x[(size_t)p * DMODEL + d];

  __shared__ float red[4];
  __shared__ float red2[4];
  __shared__ float redb[4][4];
  int w = d >> 5, lane = d & 31;

  float s = v;
  for (int m = 16; m; m >>= 1) s += __shfl_xor(s, m, 32);
  if (lane == 0) red[w] = s;
  __syncthreads();
  float mu = (red[0] + red[1] + red[2] + red[3]) * (1.0f / 128.0f);

  float dv = v - mu;
  float s2 = dv * dv;
  for (int m = 16; m; m >>= 1) s2 += __shfl_xor(s2, m, 32);
  if (lane == 0) red2[w] = s2;
  __syncthreads();
  float var = (red2[0] + red2[1] + red2[2] + red2[3]) * (1.0f / 128.0f);

  float xnv = dv * rsqrtf(var + 1e-5f) * lng[d] + lnb[d];
  int dp = transpose ? ((p % NAX) * NAX + (p / NAX)) : p;
  xn[(size_t)dp * DMODEL + d] = (bf16)xnv;

  if (doBias) {
    float b0 = v * wb[d * 4 + 0];
    float b1 = v * wb[d * 4 + 1];
    float b2 = v * wb[d * 4 + 2];
    float b3 = v * wb[d * 4 + 3];
    for (int m = 16; m; m >>= 1) {
      b0 += __shfl_xor(b0, m, 32);
      b1 += __shfl_xor(b1, m, 32);
      b2 += __shfl_xor(b2, m, 32);
      b3 += __shfl_xor(b3, m, 32);
    }
    if (lane == 0) {
      redb[w][0] = b0; redb[w][1] = b1; redb[w][2] = b2; redb[w][3] = b3;
    }
    __syncthreads();
    if (d < 4) {
      float bs = redb[0][d] + redb[1][d] + redb[2][d] + redb[3][d];
      biasb[(size_t)d * NPOS + p] = bs;
    }
  }
}

// ---------------------------------------------------------------------------
// Generic WMMA GEMM: C(M x N) = A(M x K, bf16) * B^T-stored(N x K, bf16)
// Block tile 128x64, 8 waves (4 M-waves x 2 N-waves), each wave 32x32 (2x2 WMMA).
// Ping-pong double-buffered LDS staged with global_load_async_to_lds_b128.
// mode 0: fused QKV+gate epilogue (N=512)
// mode 1: attention out-proj: +bo, +residual, optional transposed store (N=128)
// mode 2: FFN1 dual-B: t = (acc+b1[c]) * gelu(accG+b1[c+512]) -> bf16 (N=512)
// mode 3: FFN2: +b2, +residual -> f32 out (N=128, K=512)
// ---------------------------------------------------------------------------
#define BM 128
#define BN 64
#define BK 32

__global__ void __launch_bounds__(256)
gemm_kernel(const bf16* __restrict__ A, const bf16* __restrict__ B,
            const bf16* __restrict__ B2, int K, int mode, int transpose_out,
            bf16* __restrict__ oq, bf16* __restrict__ okk, bf16* __restrict__ ov,
            float* __restrict__ ogate, const float* __restrict__ bias1,
            bf16* __restrict__ obf, float* __restrict__ of,
            const float* __restrict__ resid) {
  __shared__ __align__(16) bf16 sA[2][BM * BK];
  __shared__ __align__(16) bf16 sB[2][BN * BK];
  __shared__ __align__(16) bf16 sB2[2][BN * BK];

  int tid = threadIdx.x;
  int m0 = blockIdx.x * BM;
  int n0 = blockIdx.y * BN;
  int wid = tid >> 5, lane = tid & 31;
  int wm = (wid & 3) * 32;        // wave M offset within block tile
  int wn = (wid >> 2) * 32;       // wave N offset within block tile

  v8f acc[4] = {};
  v8f accg[4] = {};

  // async-stage one K-slice (32 wide) into LDS buffer `buf`
  auto stage = [&](int buf, int k0) {
    for (int c = tid; c < (BM * BK) / 8; c += 256) {
      int rr = c >> 2, kc = (c & 3) * 8;
      async_copy_b128(A + (size_t)(m0 + rr) * K + k0 + kc,
                      lds_off(&sA[buf][rr * BK + kc]));
    }
    {
      int rr = tid >> 2, kc = (tid & 3) * 8;   // 64x32 = exactly 256 chunks
      async_copy_b128(B + (size_t)(n0 + rr) * K + k0 + kc,
                      lds_off(&sB[buf][rr * BK + kc]));
      if (mode == 2)
        async_copy_b128(B2 + (size_t)(n0 + rr) * K + k0 + kc,
                        lds_off(&sB2[buf][rr * BK + kc]));
    }
  };

  int nk = K >> 5;
  stage(0, 0);
  wait_async0();
  __syncthreads();

  for (int it = 0; it < nk; ++it) {
    int cur = it & 1;
    if (it + 1 < nk) stage(1 - cur, (it + 1) << 5);   // prefetch next slice

    v16bf af0 = load_frag(&sA[cur][(wm + 0) * BK], BK);
    v16bf af1 = load_frag(&sA[cur][(wm + 16) * BK], BK);
    v16bf bf0 = load_frag(&sB[cur][(wn + 0) * BK], BK);
    v16bf bf1 = load_frag(&sB[cur][(wn + 16) * BK], BK);
    acc[0] = wmma_bf16(af0, bf0, acc[0]);
    acc[1] = wmma_bf16(af0, bf1, acc[1]);
    acc[2] = wmma_bf16(af1, bf0, acc[2]);
    acc[3] = wmma_bf16(af1, bf1, acc[3]);
    if (mode == 2) {
      v16bf bg0 = load_frag(&sB2[cur][(wn + 0) * BK], BK);
      v16bf bg1 = load_frag(&sB2[cur][(wn + 16) * BK], BK);
      accg[0] = wmma_bf16(af0, bg0, accg[0]);
      accg[1] = wmma_bf16(af0, bg1, accg[1]);
      accg[2] = wmma_bf16(af1, bg0, accg[2]);
      accg[3] = wmma_bf16(af1, bg1, accg[3]);
    }

    if (it + 1 < nk) wait_async0();   // next slice landed
    __syncthreads();                  // everyone done reading `cur`
  }

  // epilogue: D layout -> element (m,n): lane = n + 16*(m>=8), reg = m%8
  for (int i = 0; i < 2; i++) {
    for (int j = 0; j < 2; j++) {
      v8f a = acc[i * 2 + j];
      v8f g = accg[i * 2 + j];
      for (int rr = 0; rr < 8; rr++) {
        int ml = (lane < 16) ? rr : rr + 8;
        int nl = lane & 15;
        int row = m0 + wm + i * 16 + ml;
        int col = n0 + wn + j * 16 + nl;
        float v = a[rr];
        if (mode == 0) {
          if (col < 128) {
            oq[(size_t)row * 128 + col] = (bf16)(v * 0.17677669529663687f);
          } else if (col < 256) {
            okk[(size_t)row * 128 + (col - 128)] = (bf16)v;
          } else if (col < 384) {
            ov[(size_t)row * 128 + (col - 256)] = (bf16)v;
          } else {
            float z = v + bias1[col - 384];
            ogate[(size_t)row * 128 + (col - 384)] = 1.0f / (1.0f + expf(-z));
          }
        } else if (mode == 1 || mode == 3) {
          int orow = row;
          if (transpose_out) orow = (row % NAX) * NAX + (row / NAX);
          size_t oi = (size_t)orow * 128 + col;
          of[oi] = resid[oi] + v + bias1[col];
        } else {  // mode 2
          float aa = v + bias1[col];
          float gg = g[rr] + bias1[col + 512];
          float gel = 0.5f * gg * (1.0f + erff(gg * 0.70710678118f));
          obf[(size_t)row * 512 + col] = (bf16)(aa * gel);
        }
      }
    }
  }
}

// ---------------------------------------------------------------------------
// Attention kernel: one block per (q-tile of 32, r, h). 256 threads = 8 waves.
// scores = q.k^T + bias ; softmax over k (192) ; o = P.v ; o *= gates.
// K/Q tiles staged via async global->LDS; V staged transposed via LDS stores.
// ---------------------------------------------------------------------------
__global__ void __launch_bounds__(256)
attn_kernel(const bf16* __restrict__ qb, const bf16* __restrict__ kb,
            const bf16* __restrict__ vb, const float* __restrict__ gates,
            const float* __restrict__ biasb, bf16* __restrict__ og) {
  int qt = blockIdx.x;  // 0..5 (32-query tiles)
  int r  = blockIdx.y;  // 0..191
  int h  = blockIdx.z;  // 0..3

  __shared__ __align__(16) bf16 sQ[32 * 32];
  __shared__ __align__(16) bf16 sK[192 * 32];
  __shared__ __align__(16) bf16 sVT[32 * 192];
  __shared__ __align__(16) float sS[32 * 192];
  __shared__ __align__(16) bf16 sP[32 * 192];

  int tid = threadIdx.x;
  int wid = tid >> 5, lane = tid & 31;
  int q0 = qt * 32;

  // async-stage K (192 x 32): 768 chunks of 8 bf16
  for (int c = tid; c < 768; c += 256) {
    int kk = c >> 2, e0 = (c & 3) * 8;
    async_copy_b128(kb + (size_t)(r * NAX + kk) * 128 + h * 32 + e0,
                    lds_off(&sK[kk * 32 + e0]));
  }
  // async-stage Q tile (32 x 32): 128 chunks
  if (tid < 128) {
    int qq = tid >> 2, e0 = (tid & 3) * 8;
    async_copy_b128(qb + (size_t)(r * NAX + q0 + qq) * 128 + h * 32 + e0,
                    lds_off(&sQ[qq * 32 + e0]));
  }
  // stage V transposed (32 x 192): scalar (transpose on the way in)
  for (int idx = tid; idx < 6144; idx += 256) {
    int kk = idx >> 5, e = idx & 31;
    sVT[e * 192 + kk] = vb[(size_t)(r * NAX + kk) * 128 + h * 32 + e];
  }
  wait_async0();
  __syncthreads();

  // scores: waves arranged 2(M) x 4(N); each wave owns 16 q-rows x 48 k-cols
  {
    int wm = (wid & 1) * 16;
    int wn = (wid >> 1) * 48;
    v16bf aq = load_frag(sQ + wm * 32, 32);
    for (int nt = 0; nt < 3; nt++) {
      int ncol = wn + nt * 16;
      v16bf bk = load_frag(sK + ncol * 32, 32);
      v8f sc = {};
      sc = wmma_bf16(aq, bk, sc);
      for (int rr = 0; rr < 8; rr++) {
        int ml = (lane < 16) ? rr : rr + 8;
        int nl = lane & 15;
        int qpos = q0 + wm + ml;
        int kcol = ncol + nl;
        sS[(wm + ml) * 192 + kcol] =
            sc[rr] + biasb[(size_t)h * NPOS + qpos * NAX + kcol];
      }
    }
  }
  __syncthreads();

  // softmax over 192: 8 threads per row (rows 0..31)
  {
    int row = tid >> 3, sub = tid & 7;
    float mx = -1e30f;
    for (int c = sub; c < 192; c += 8) mx = fmaxf(mx, sS[row * 192 + c]);
    for (int m = 1; m < 8; m <<= 1) mx = fmaxf(mx, __shfl_xor(mx, m, 32));
    float sum = 0.0f;
    for (int c = sub; c < 192; c += 8) sum += expf(sS[row * 192 + c] - mx);
    for (int m = 1; m < 8; m <<= 1) sum += __shfl_xor(sum, m, 32);
    float inv = 1.0f / sum;
    for (int c = sub; c < 192; c += 8)
      sP[row * 192 + c] = (bf16)(expf(sS[row * 192 + c] - mx) * inv);
  }
  __syncthreads();

  // o = P(32x192) . v(192x32): 4 waves, each one 16x16 tile, K in 6 steps
  if (wid < 4) {
    int wm = (wid & 1) * 16;
    int wn = (wid >> 1) * 16;
    v8f o = {};
    for (int kk0 = 0; kk0 < 192; kk0 += 32) {
      v16bf ap = load_frag(sP + wm * 192 + kk0, 192);
      v16bf bv = load_frag(sVT + wn * 192 + kk0, 192);
      o = wmma_bf16(ap, bv, o);
    }
    for (int rr = 0; rr < 8; rr++) {
      int ml = (lane < 16) ? rr : rr + 8;
      int nl = lane & 15;
      int qpos = q0 + wm + ml;
      int e = wn + nl;
      size_t p = (size_t)(r * NAX + qpos) * 128 + h * 32 + e;
      og[p] = (bf16)(o[rr] * gates[p]);
    }
  }
}

// ---------------------------------------------------------------------------
// Host launcher
// ---------------------------------------------------------------------------
extern "C" void kernel_launch(void* const* d_in, const int* in_sizes, int n_in,
                              void* d_out, int out_size, void* d_ws, size_t ws_size,
                              hipStream_t stream) {
  (void)in_sizes; (void)n_in; (void)out_size; (void)ws_size;

  const float* x_in  = (const float*)d_in[0];
  const float* lng_o = (const float*)d_in[1];
  const float* lnb_o = (const float*)d_in[2];
  const float* wq_o  = (const float*)d_in[3];
  const float* wkv_o = (const float*)d_in[4];
  const float* wo_o  = (const float*)d_in[5];
  const float* bo_o  = (const float*)d_in[6];
  const float* wg_o  = (const float*)d_in[7];
  const float* bg_o  = (const float*)d_in[8];
  const float* wb_o  = (const float*)d_in[9];
  const float* lng_i = (const float*)d_in[10];
  const float* lnb_i = (const float*)d_in[11];
  const float* wq_i  = (const float*)d_in[12];
  const float* wkv_i = (const float*)d_in[13];
  const float* wo_i  = (const float*)d_in[14];
  const float* bo_i  = (const float*)d_in[15];
  const float* wg_i  = (const float*)d_in[16];
  const float* bg_i  = (const float*)d_in[17];
  const float* wb_i  = (const float*)d_in[18];
  const float* lng_f = (const float*)d_in[19];
  const float* lnb_f = (const float*)d_in[20];
  const float* w1    = (const float*)d_in[21];
  const float* b1    = (const float*)d_in[22];
  const float* w2    = (const float*)d_in[23];
  const float* b2    = (const float*)d_in[24];
  float* out = (float*)d_out;

  // workspace carve-up (256B aligned regions)
  char* ws = (char*)d_ws;
  size_t off = 0;
  auto carve = [&](size_t bytes) {
    void* p = ws + off;
    off = (off + bytes + 255) & ~(size_t)255;
    return p;
  };
  bf16* WcatO = (bf16*)carve(512 * 128 * sizeof(bf16));
  bf16* WoTO  = (bf16*)carve(128 * 128 * sizeof(bf16));
  bf16* WcatI = (bf16*)carve(512 * 128 * sizeof(bf16));
  bf16* WoTI  = (bf16*)carve(128 * 128 * sizeof(bf16));
  bf16* W1T   = (bf16*)carve(1024 * 128 * sizeof(bf16));
  bf16* W2T   = (bf16*)carve(128 * 512 * sizeof(bf16));
  bf16* XN    = (bf16*)carve((size_t)NPOS * 128 * sizeof(bf16));
  bf16* Qb    = (bf16*)carve((size_t)NPOS * 128 * sizeof(bf16));
  bf16* Kb    = (bf16*)carve((size_t)NPOS * 128 * sizeof(bf16));
  bf16* Vb    = (bf16*)carve((size_t)NPOS * 128 * sizeof(bf16));
  float* Gates = (float*)carve((size_t)NPOS * 128 * sizeof(float));
  float* BiasB = (float*)carve((size_t)4 * NPOS * sizeof(float));
  bf16* OG    = (bf16*)carve((size_t)NPOS * 128 * sizeof(bf16));
  bf16* TMP   = (bf16*)carve((size_t)NPOS * 512 * sizeof(bf16));

  auto wconv = [&](const float* w, bf16* wt, int R, int C) {
    int n = R * C;
    wconv_kernel<<<(n + 255) / 256, 256, 0, stream>>>(w, wt, R, C);
  };
  // bf16 B^T weights (N x K layouts)
  wconv(wq_o,  WcatO,             128, 128);
  wconv(wkv_o, WcatO + 128 * 128, 128, 256);
  wconv(wg_o,  WcatO + 384 * 128, 128, 128);
  wconv(wo_o,  WoTO,              128, 128);
  wconv(wq_i,  WcatI,             128, 128);
  wconv(wkv_i, WcatI + 128 * 128, 128, 256);
  wconv(wg_i,  WcatI + 384 * 128, 128, 128);
  wconv(wo_i,  WoTI,              128, 128);
  wconv(w1,    W1T,               128, 1024);
  wconv(w2,    W2T,               512, 128);

  dim3 gQKVG(NPOS / BM, 512 / BN);
  dim3 gPROJ(NPOS / BM, 128 / BN);
  dim3 gFFN1(NPOS / BM, 512 / BN);
  dim3 gATTN(6, NAX, NHEAD);

  // ---- pass 1: row-axial ----
  ln_kernel<<<NPOS, 128, 0, stream>>>(x_in, lng_o, lnb_o, wb_o, XN, BiasB,
                                      /*transpose=*/0, /*doBias=*/1);
  gemm_kernel<<<gQKVG, 256, 0, stream>>>(XN, WcatO, nullptr, 128, /*mode=*/0, 0,
                                         Qb, Kb, Vb, Gates, bg_o,
                                         nullptr, nullptr, nullptr);
  attn_kernel<<<gATTN, 256, 0, stream>>>(Qb, Kb, Vb, Gates, BiasB, OG);
  gemm_kernel<<<gPROJ, 256, 0, stream>>>(OG, WoTO, nullptr, 128, /*mode=*/1, 0,
                                         nullptr, nullptr, nullptr, nullptr, bo_o,
                                         nullptr, out, x_in);

  // ---- pass 2: column-axial (activations pre-permuted, output de-permuted) ----
  ln_kernel<<<NPOS, 128, 0, stream>>>(out, lng_i, lnb_i, wb_i, XN, BiasB,
                                      /*transpose=*/1, /*doBias=*/1);
  gemm_kernel<<<gQKVG, 256, 0, stream>>>(XN, WcatI, nullptr, 128, /*mode=*/0, 0,
                                         Qb, Kb, Vb, Gates, bg_i,
                                         nullptr, nullptr, nullptr);
  attn_kernel<<<gATTN, 256, 0, stream>>>(Qb, Kb, Vb, Gates, BiasB, OG);
  gemm_kernel<<<gPROJ, 256, 0, stream>>>(OG, WoTI, nullptr, 128, /*mode=*/1,
                                         /*transpose_out=*/1,
                                         nullptr, nullptr, nullptr, nullptr, bo_i,
                                         nullptr, out, out);

  // ---- FFN (GEGLU) ----
  ln_kernel<<<NPOS, 128, 0, stream>>>(out, lng_f, lnb_f, nullptr, XN, nullptr,
                                      /*transpose=*/0, /*doBias=*/0);
  gemm_kernel<<<gFFN1, 256, 0, stream>>>(XN, W1T, W1T + 512 * 128, 128,
                                         /*mode=*/2, 0,
                                         nullptr, nullptr, nullptr, nullptr, b1,
                                         TMP, nullptr, nullptr);
  gemm_kernel<<<gPROJ, 256, 0, stream>>>(TMP, W2T, nullptr, 512, /*mode=*/3, 0,
                                         nullptr, nullptr, nullptr, nullptr, b2,
                                         nullptr, out, out);
}
